// AdaptiveAttention_5514738008373
// MI455X (gfx1250) — matrix operand order repaired and checked by reading, not compiled, not run on hardware
//
#include <hip/hip_runtime.h>
#include <hip/hip_bf16.h>

#define HID   512
#define G4    2048   // 4*HID
#define GSN   2560   // 4*HID + HID  (gates + s_pre)
#define BB    32
#define TT    64
#define BT    2048   // B*T
#define KATT  49
#define VOCAB 10000
#define VPAD  10240  // 640 tiles of 16 -> 160 groups of 4

typedef __attribute__((ext_vector_type(16))) __bf16 v16bf;
typedef __attribute__((ext_vector_type(8)))  float  v8f;

union Frag16 { unsigned short u[16]; v16bf v; };

__device__ __forceinline__ unsigned short f2bf(float f) {
    unsigned int u = __float_as_uint(f);
    unsigned int r = u + 0x7FFFu + ((u >> 16) & 1u);   // round-to-nearest-even
    return (unsigned short)(r >> 16);
}

__device__ __forceinline__ float sigm(float x) {
    return 1.0f / (1.0f + __expf(-x));
}

__device__ __forceinline__ void load_a(Frag16& a, const unsigned short* __restrict__ ap, int kk) {
#pragma unroll
    for (int i = 0; i < 8; i++) { a.u[i] = ap[kk + i]; a.u[8 + i] = ap[kk + 16 + i]; }
}
__device__ __forceinline__ void load_b(Frag16& b, const unsigned short* __restrict__ bp, int kk) {
#pragma unroll
    for (int i = 0; i < 8; i++) { b.u[i] = bp[kk + i]; b.u[8 + i] = bp[kk + 16 + i]; }
}

// ---------------------------------------------------------------------------
// Single 16x16 tile accumulate (latency-critical recurrent GEMM).
// Arow = A + tile_m*16*K (A: MxK row-major bf16); Brow = W + tile_n*16*K
// (W: NxK row-major bf16, so B[k][n] = W[n][k]).
// ---------------------------------------------------------------------------
__device__ __forceinline__ v8f wmma_tile_acc(const unsigned short* __restrict__ Arow,
                                             const unsigned short* __restrict__ Brow,
                                             int K, int lane) {
    v8f acc = {};
    const int half = lane >> 4;
    const int mr   = lane & 15;
    const unsigned short* ap = Arow + (size_t)mr * K + half * 8;
    const unsigned short* bp = Brow + (size_t)mr * K + half * 8;
    for (int kk = 0; kk < K; kk += 32) {
        Frag16 a, b;
        load_a(a, ap, kk);
        load_b(b, bp, kk);
        acc = __builtin_amdgcn_wmma_f32_16x16x32_bf16(false, a.v, false, b.v,
                                                      (short)0, acc, false, false);
    }
    return acc;
}

// ---------------------------------------------------------------------------
// 16x64 register tile per wave, explicitly double-buffered (ping-pong):
// while WMMAs consume fragment set 0, the 10 b128 loads of set 1 are in
// flight -> no s_wait_loadcnt 0x0 stalls, and A traffic is cut 4x.
// Requires K % 64 == 0 (K = 512 here).
// ---------------------------------------------------------------------------
__device__ __forceinline__ void wmma_tile_acc4(const unsigned short* __restrict__ Arow,
                                               const unsigned short* __restrict__ Brow,
                                               int K, int lane, v8f acc[4]) {
    const int half = lane >> 4;
    const int mr   = lane & 15;
    const unsigned short* ap = Arow + (size_t)mr * K + half * 8;
    const unsigned short* bp = Brow + (size_t)mr * K + half * 8;

    Frag16 a0, a1, b0[4], b1[4];
    load_a(a0, ap, 0);
#pragma unroll
    for (int j = 0; j < 4; j++) load_b(b0[j], bp + (size_t)j * 16 * K, 0);
    acc[0] = v8f{}; acc[1] = v8f{}; acc[2] = v8f{}; acc[3] = v8f{};

    for (int kk = 0; kk < K; kk += 64) {
        // prefetch k-slice kk+32 into set 1
        load_a(a1, ap, kk + 32);
#pragma unroll
        for (int j = 0; j < 4; j++) load_b(b1[j], bp + (size_t)j * 16 * K, kk + 32);
        // consume set 0
#pragma unroll
        for (int j = 0; j < 4; j++)
            acc[j] = __builtin_amdgcn_wmma_f32_16x16x32_bf16(false, a0.v, false, b0[j].v,
                                                             (short)0, acc[j], false, false);
        // prefetch k-slice kk+64 into set 0 (tail reloads kk=0, discarded)
        const int kn = (kk + 64 < K) ? kk + 64 : 0;
        load_a(a0, ap, kn);
#pragma unroll
        for (int j = 0; j < 4; j++) load_b(b0[j], bp + (size_t)j * 16 * K, kn);
        // consume set 1
#pragma unroll
        for (int j = 0; j < 4; j++)
            acc[j] = __builtin_amdgcn_wmma_f32_16x16x32_bf16(false, a1.v, false, b1[j].v,
                                                             (short)0, acc[j], false, false);
    }
}

// ---------------------------------------------------------------------------
// Utility kernels
// ---------------------------------------------------------------------------
__global__ void k_cvt_bf16(const float* __restrict__ src, unsigned short* __restrict__ dst, int n) {
    int i = blockIdx.x * 256 + threadIdx.x;
    if (i < n) dst[i] = f2bf(src[i]);
}
__global__ void k_zero_u16(unsigned short* __restrict__ p, int n) {
    int i = blockIdx.x * 256 + threadIdx.x;
    if (i < n) p[i] = 0;
}
__global__ void k_zero_f32(float* __restrict__ p, int n) {
    int i = blockIdx.x * 256 + threadIdx.x;
    if (i < n) p[i] = 0.0f;
}

// ---------------------------------------------------------------------------
// x_gates = x @ Wih^T + (bih + bhh)   [2048 x 2048], K=512, 4-way N-blocked
// ---------------------------------------------------------------------------
__global__ __launch_bounds__(256) void k_gemm_xgates(
    const unsigned short* __restrict__ A, const unsigned short* __restrict__ W,
    const float* __restrict__ bih, const float* __restrict__ bhh,
    float* __restrict__ out) {
    const int lane = threadIdx.x & 31;
    const int gw = blockIdx.x * 8 + (threadIdx.x >> 5);
    const int TG = (G4 / 16) / 4;            // 32 groups of 4 N-tiles
    const int tm = gw / TG, tg = gw % TG;
    v8f acc[4];
    wmma_tile_acc4(A + (size_t)tm * 16 * HID, W + (size_t)(tg * 4) * 16 * HID, HID, lane, acc);
    const int m0 = tm * 16 + (lane >> 4) * 8;
#pragma unroll
    for (int j = 0; j < 4; j++) {
        const int n = (tg * 4 + j) * 16 + (lane & 15);
        const float bias = bih[n] + bhh[n];
#pragma unroll
        for (int r = 0; r < 8; r++) out[(size_t)(m0 + r) * G4 + n] = acc[j][r] + bias;
    }
}

// ---------------------------------------------------------------------------
// x_sent = x @ Wx_s^T    [2048 x 512], K=512, 4-way N-blocked
// ---------------------------------------------------------------------------
__global__ __launch_bounds__(256) void k_gemm_xsent(
    const unsigned short* __restrict__ A, const unsigned short* __restrict__ W,
    float* __restrict__ out) {
    const int lane = threadIdx.x & 31;
    const int gw = blockIdx.x * 8 + (threadIdx.x >> 5);
    const int TG = (HID / 16) / 4;           // 8 groups
    const int tm = gw / TG, tg = gw % TG;
    v8f acc[4];
    wmma_tile_acc4(A + (size_t)tm * 16 * HID, W + (size_t)(tg * 4) * 16 * HID, HID, lane, acc);
    const int m0 = tm * 16 + (lane >> 4) * 8;
#pragma unroll
    for (int j = 0; j < 4; j++) {
        const int n = (tg * 4 + j) * 16 + (lane & 15);
#pragma unroll
        for (int r = 0; r < 8; r++) out[(size_t)(m0 + r) * HID + n] = acc[j][r];
    }
}

// ---------------------------------------------------------------------------
// Per-step recurrent GEMM (critical path -> unblocked for max parallelism):
//   gs[b][n] = h_prev@[Whh;Wh_s]^T + (n<2048 ? x_gates[b,t,n] : x_sent[b,t,n-2048])
//   [32 x 2560], K=512 -> 320 tiles -> 40 blocks
// ---------------------------------------------------------------------------
__global__ __launch_bounds__(256) void k_gemm_step(
    const unsigned short* __restrict__ hbf, const unsigned short* __restrict__ Wcomb,
    const float* __restrict__ xg, const float* __restrict__ xs,
    float* __restrict__ gs, int t) {
    const int lane = threadIdx.x & 31;
    const int gw = blockIdx.x * 8 + (threadIdx.x >> 5);
    const int TN = GSN / 16;                 // 160
    const int tm = gw / TN, tn = gw % TN;
    v8f acc = wmma_tile_acc(hbf + (size_t)tm * 16 * HID, Wcomb + (size_t)tn * 16 * HID, HID, lane);
    const int n  = tn * 16 + (lane & 15);
    const int m0 = tm * 16 + (lane >> 4) * 8;
#pragma unroll
    for (int r = 0; r < 8; r++) {
        const int m = m0 + r;   // batch index
        float bias = (n < G4) ? xg[((size_t)m * TT + t) * G4 + n]
                              : xs[((size_t)m * TT + t) * HID + (n - G4)];
        gs[(size_t)m * GSN + n] = acc[r] + bias;
    }
}

// ---------------------------------------------------------------------------
// Fused LSTM cell + adaptive attention, one block per batch item.
// ---------------------------------------------------------------------------
__global__ __launch_bounds__(256) void k_cell_attn(
    const float* __restrict__ gs,            // [32][2560]
    float* __restrict__ cbuf,                // [32][512]
    unsigned short* __restrict__ hbf,        // [32][512]
    const float* __restrict__ V,             // [32][49][512]
    const float* __restrict__ Vk,            // [32][49][49]
    const float* __restrict__ Wg,            // [49][512]
    const float* __restrict__ Ws,            // [49][512]
    const float* __restrict__ watt,          // [49]
    unsigned short* __restrict__ opbf,       // [2048][512]
    int t) {
    const int b = blockIdx.x, tid = threadIdx.x;
    __shared__ float sh_h[HID], sh_s[HID];
    __shared__ float sh_hg[KATT], sh_sw[KATT], sh_z[KATT + 1], sh_alpha[KATT];
    __shared__ float sh_beta;

    const float* g = gs + (size_t)b * GSN;
    for (int j = tid; j < HID; j += 256) {
        float ig = g[j], fg = g[HID + j], gg = g[2 * HID + j], og = g[3 * HID + j];
        float cp = cbuf[b * HID + j];
        float cn = sigm(fg) * cp + sigm(ig) * tanhf(gg);
        float hv = sigm(og) * tanhf(cn);
        cbuf[b * HID + j] = cn;
        sh_h[j] = hv;
        hbf[b * HID + j] = f2bf(hv);
        sh_s[j] = sigm(g[4 * HID + j]) * tanhf(cn);   // s = sigmoid(xs + h_prev@Wh_s^T) * tanh(c_new)
    }
    __syncthreads();

    // hg = h @ Wg^T, sw = s @ Ws^T  (49 dots of 512, one k per wave, round-robin)
    const int wave = tid >> 5, lane = tid & 31;
    for (int k = wave; k < KATT; k += 8) {
        const float* wgr = Wg + (size_t)k * HID;
        const float* wsr = Ws + (size_t)k * HID;
        float a1 = 0.0f, a2 = 0.0f;
        for (int j = lane; j < HID; j += 32) { a1 += wgr[j] * sh_h[j]; a2 += wsr[j] * sh_s[j]; }
#pragma unroll
        for (int off = 16; off > 0; off >>= 1) {
            a1 += __shfl_down(a1, off);
            a2 += __shfl_down(a2, off);
        }
        if (lane == 0) { sh_hg[k] = a1; sh_sw[k] = a2; }
    }
    __syncthreads();

    if (tid < KATT) {
        const float* vk = Vk + ((size_t)b * KATT + tid) * KATT;
        float z = 0.0f;
        for (int k = 0; k < KATT; k++) z += tanhf(vk[k] + sh_hg[k]) * watt[k];
        sh_z[tid] = z;
    } else if (tid == KATT) {
        float hw = 0.0f;
        for (int k = 0; k < KATT; k++) hw += tanhf(sh_sw[k] + sh_hg[k]) * watt[k];
        sh_z[KATT] = hw;
    }
    __syncthreads();

    if (tid == 0) {
        float m1 = -1e30f;
        for (int l = 0; l < KATT; l++) m1 = fmaxf(m1, sh_z[l]);
        float s1 = 0.0f;
        for (int l = 0; l < KATT; l++) { float e = __expf(sh_z[l] - m1); sh_alpha[l] = e; s1 += e; }
        float inv = 1.0f / s1;
        for (int l = 0; l < KATT; l++) sh_alpha[l] *= inv;
        const float how = sh_z[KATT];
        const float m2 = fmaxf(m1, how);
        const float den = s1 * __expf(m1 - m2) + __expf(how - m2);
        sh_beta = __expf(how - m2) / den;
    }
    __syncthreads();

    const float beta = sh_beta;
    for (int j = tid; j < HID; j += 256) {
        const float* vb = V + (size_t)b * KATT * HID + j;
        float ct = 0.0f;
        for (int l = 0; l < KATT; l++) ct += sh_alpha[l] * vb[(size_t)l * HID];
        float chat = beta * sh_s[j] + (1.0f - beta) * ct;
        opbf[((size_t)b * TT + t) * HID + j] = f2bf(chat + sh_h[j]);
    }
}

// ---------------------------------------------------------------------------
// Vk[b][l][k] = sum_h V[b][l][h] * Wv[k][h]   (tiny, fp32 scalar)
// ---------------------------------------------------------------------------
__global__ __launch_bounds__(256) void k_vk(const float* __restrict__ V,
                                            const float* __restrict__ Wv,
                                            float* __restrict__ Vk) {
    const int b = blockIdx.x, tid = threadIdx.x;
    for (int idx = tid; idx < KATT * KATT; idx += 256) {
        const int l = idx / KATT, k = idx % KATT;
        const float* vr = V + ((size_t)b * KATT + l) * HID;
        const float* wr = Wv + (size_t)k * HID;
        float acc = 0.0f;
        for (int j = 0; j < 512; j++) acc += vr[j] * wr[j];
        Vk[(size_t)b * KATT * KATT + idx] = acc;
    }
}

// ---------------------------------------------------------------------------
// y = op @ Wmlp^T + bmlp   [2048 x 10000], K=512, N padded to 10240, 4-way
// N-blocked + double-buffered (dominant GEMM: 21 GFLOP, L2-resident weights)
// ---------------------------------------------------------------------------
__global__ __launch_bounds__(256) void k_gemm_vocab(
    const unsigned short* __restrict__ A, const unsigned short* __restrict__ W,
    const float* __restrict__ bmlp, float* __restrict__ out) {
    const int lane = threadIdx.x & 31;
    const int gw = blockIdx.x * 8 + (threadIdx.x >> 5);
    const int TG = (VPAD / 16) / 4;          // 160 groups
    const int tm = gw / TG, tg = gw % TG;
    v8f acc[4];
    wmma_tile_acc4(A + (size_t)tm * 16 * HID, W + (size_t)(tg * 4) * 16 * HID, HID, lane, acc);
    const int m0 = tm * 16 + (lane >> 4) * 8;
#pragma unroll
    for (int j = 0; j < 4; j++) {
        const int n = (tg * 4 + j) * 16 + (lane & 15);
        if (n < VOCAB) {
            const float bias = bmlp[n];
#pragma unroll
            for (int r = 0; r < 8; r++) out[(size_t)(m0 + r) * VOCAB + n] = acc[j][r] + bias;
        }
    }
}

// ---------------------------------------------------------------------------
extern "C" void kernel_launch(void* const* d_in, const int* in_sizes, int n_in,
                              void* d_out, int out_size, void* d_ws, size_t ws_size,
                              hipStream_t stream) {
    const float* x     = (const float*)d_in[0];
    const float* V     = (const float*)d_in[1];
    const float* Wih   = (const float*)d_in[2];
    const float* Whh   = (const float*)d_in[3];
    const float* bih   = (const float*)d_in[4];
    const float* bhh   = (const float*)d_in[5];
    const float* Wx_s  = (const float*)d_in[6];
    const float* Wh_s  = (const float*)d_in[7];
    const float* Wv    = (const float*)d_in[8];
    const float* Wg    = (const float*)d_in[9];
    const float* Watt  = (const float*)d_in[10];
    const float* Ws    = (const float*)d_in[11];
    const float* Wmlp  = (const float*)d_in[12];
    const float* bmlp  = (const float*)d_in[13];
    float* out = (float*)d_out;

    // ---- deterministic workspace carve-up (~42 MB) ----
    char* ws = (char*)d_ws;
    size_t off = 0;
    auto carve = [&](size_t bytes) -> char* {
        char* p = ws + off;
        off += (bytes + 255) & ~(size_t)255;
        return p;
    };
    unsigned short* xbf    = (unsigned short*)carve((size_t)BT * HID * 2);
    unsigned short* wihbf  = (unsigned short*)carve((size_t)G4 * HID * 2);
    unsigned short* wcomb  = (unsigned short*)carve((size_t)GSN * HID * 2);   // [Whh ; Wh_s]
    unsigned short* wxsbf  = (unsigned short*)carve((size_t)HID * HID * 2);
    unsigned short* wmlpbf = (unsigned short*)carve((size_t)VPAD * HID * 2);
    unsigned short* opbf   = (unsigned short*)carve((size_t)BT * HID * 2);
    unsigned short* hbf    = (unsigned short*)carve((size_t)BB * HID * 2);
    float* xg   = (float*)carve((size_t)BT * G4 * 4);
    float* xs   = (float*)carve((size_t)BT * HID * 4);
    float* vk   = (float*)carve((size_t)BB * KATT * KATT * 4);
    float* cbuf = (float*)carve((size_t)BB * HID * 4);
    float* gs   = (float*)carve((size_t)BB * GSN * 4);
    (void)ws_size; (void)n_in; (void)in_sizes; (void)out_size;

    auto blocks = [](int n) { return (n + 255) / 256; };

    // ---- precision conversion (fp32 -> bf16) ----
    k_cvt_bf16<<<blocks(BT * HID), 256, 0, stream>>>(x, xbf, BT * HID);
    k_cvt_bf16<<<blocks(G4 * HID), 256, 0, stream>>>(Wih, wihbf, G4 * HID);
    k_cvt_bf16<<<blocks(G4 * HID), 256, 0, stream>>>(Whh, wcomb, G4 * HID);
    k_cvt_bf16<<<blocks(HID * HID), 256, 0, stream>>>(Wh_s, wcomb + (size_t)G4 * HID, HID * HID);
    k_cvt_bf16<<<blocks(HID * HID), 256, 0, stream>>>(Wx_s, wxsbf, HID * HID);
    k_cvt_bf16<<<blocks(VOCAB * HID), 256, 0, stream>>>(Wmlp, wmlpbf, VOCAB * HID);
    k_zero_u16<<<blocks((VPAD - VOCAB) * HID), 256, 0, stream>>>(
        wmlpbf + (size_t)VOCAB * HID, (VPAD - VOCAB) * HID);

    // ---- initial state h0 = c0 = 0 ----
    k_zero_f32<<<blocks(BB * HID), 256, 0, stream>>>(cbuf, BB * HID);
    k_zero_u16<<<blocks(BB * HID), 256, 0, stream>>>(hbf, BB * HID);

    // ---- batched input projections (WMMA bf16, 4-way N-blocked) ----
    k_gemm_xgates<<<(BT / 16) * ((G4 / 16) / 4) / 8, 256, 0, stream>>>(xbf, wihbf, bih, bhh, xg);
    k_gemm_xsent<<<(BT / 16) * ((HID / 16) / 4) / 8, 256, 0, stream>>>(xbf, wxsbf, xs);
    k_vk<<<BB, 256, 0, stream>>>(V, Wv, vk);

    // ---- sequential scan: 64 steps, 2 kernels per step ----
    for (int t = 0; t < TT; t++) {
        k_gemm_step<<<(2 * (GSN / 16)) / 8, 256, 0, stream>>>(hbf, wcomb, xg, xs, gs, t);
        k_cell_attn<<<BB, 256, 0, stream>>>(gs, cbuf, hbf, V, vk, Wg, Ws, Watt, opbf, t);
    }

    // ---- deferred vocab projection: dominant GEMM, once (WMMA bf16) ----
    k_gemm_vocab<<<(BT / 16) * ((VPAD / 16) / 4) / 8, 256, 0, stream>>>(opbf, wmlpbf, bmlp, out);
}
